// LLaDA2MoeGate_9191230013599
// MI455X (gfx1250) — matrix operand order repaired and checked by reading, not compiled, not run on hardware
//
#include <hip/hip_runtime.h>
#include <hip/hip_bf16.h>

typedef float  v2f   __attribute__((ext_vector_type(2)));
typedef float  v8f   __attribute__((ext_vector_type(8)));
typedef __bf16 v8bf  __attribute__((ext_vector_type(8)));
typedef __bf16 v16bf __attribute__((ext_vector_type(16)));

#define HIDDEN      2048
#define NUM_EXPERTS 256
#define N_GROUP     8
#define EPG         (NUM_EXPERTS / N_GROUP)   // 32 experts per group
#define TOPK_GROUP  4
#define TOP_K       8
#define ROUTED_SCALING 2.5f

#define TOK_TILE    32          // tokens per block (2 x 16-row WMMA tiles)
#define KC          128         // K-chunk staged in LDS
#define RS          136         // padded LDS row stride (bf16 elems) for A planes
#define SC_STRIDE   257         // padded LDS row stride (floats) for score tile

union bffrag { v16bf v; v8bf h[2]; };

// ---------------------------------------------------------------------------
// Pre-pass: split fp32 weights into bf16 hi/lo planes (Markidis decomposition)
// w = hi + lo with hi = bf16(w), lo = bf16(w - hi).  2 MB out, L2-resident.
// ---------------------------------------------------------------------------
__global__ __launch_bounds__(512)
void split_weight_kernel(const float* __restrict__ w,
                         __bf16* __restrict__ w_hi,
                         __bf16* __restrict__ w_lo, int n)
{
    const int i = blockIdx.x * blockDim.x + threadIdx.x;
    if (i < n) {
        const float x = w[i];
        const __bf16 h = (__bf16)x;
        w_hi[i] = h;
        w_lo[i] = (__bf16)(x - (float)h);
    }
}

// ---------------------------------------------------------------------------
// Main kernel. Block = 512 threads = 16 waves (wave32). Wave w computes TWO
// 16x16 logits tiles (tokens [0,16) / [16,32)) for experts [16w,16w+16).
// fp32 GEMM emulated with 3 bf16 WMMAs per K=32 (hi*hi + hi*lo + lo*hi),
// fp32 accumulation in V_WMMA_F32_16X16X32_BF16: 2.7x fewer matrix ops than
// the K=4 fp32 WMMA path on a pipe that is 8x wider per instruction.
// ---------------------------------------------------------------------------
__global__ __launch_bounds__(512)
void moe_gate_kernel(const float*  __restrict__ hs,      // (N, 2048) f32
                     const __bf16* __restrict__ w_hi,    // (256, 2048) bf16
                     const __bf16* __restrict__ w_lo,    // (256, 2048) bf16
                     const float*  __restrict__ bias,    // (256,)
                     float* __restrict__ out_idx,        // (N, 8) indices as float
                     float* __restrict__ out_w,          // (N, 8)
                     float* __restrict__ out_logits)     // (N, 256)
{
    __shared__ __bf16 AsH[TOK_TILE * RS];     // 32 x 128 A chunk, hi plane
    __shared__ __bf16 AsL[TOK_TILE * RS];     // 32 x 128 A chunk, lo plane
    __shared__ float  Sc[TOK_TILE * SC_STRIDE];
    __shared__ float  Bl[NUM_EXPERTS];

    const int tid    = threadIdx.x;
    const int lane   = tid & 31;
    const int wave   = tid >> 5;                          // 0..15 -> expert tile
    const int token0 = blockIdx.x * TOK_TILE;

    if (tid < NUM_EXPERTS) Bl[tid] = bias[tid];

    // Per-lane fragment coordinates (ISA 16-bit WMMA layouts, wave32):
    //   A 16x32: row = lane%16; K runs {kb..kb+7} and {kb+16..kb+23}, kb = 8*(lane/16)
    //   B 32x16: col = lane%16; K runs {kb..kb+7} and {kb+16..kb+23}, kb = 8*(lane/16)
    const int arow   = lane & 15;
    const int kbA    = (lane >> 4) * 8;
    const int koffB  = (lane >> 4) * 8;
    const int expert = wave * 16 + (lane & 15);

    const __bf16* bhp  = w_hi + (size_t)expert * HIDDEN + koffB;
    const __bf16* blp  = w_lo + (size_t)expert * HIDDEN + koffB;
    const __bf16* a0hp = &AsH[arow * RS + kbA];            // tokens 0..15, hi
    const __bf16* a0lp = &AsL[arow * RS + kbA];            // tokens 0..15, lo
    const __bf16* a1hp = a0hp + 16 * RS;                   // tokens 16..31, hi
    const __bf16* a1lp = a0lp + 16 * RS;                   // tokens 16..31, lo

    v8f c0 = {};
    v8f c1 = {};

    for (int kc = 0; kc < HIDDEN; kc += KC) {
        // --- cooperative stage + fp32->bf16 hi/lo split: 32x128 floats,
        //     8 contiguous K per thread (two float4 loads, two b128 LDS stores)
        {
            const int e   = tid * 8;          // 0..4095
            const int row = e >> 7;           // / 128
            const int col = e & 127;
            const float* gp = hs + (size_t)(token0 + row) * HIDDEN + kc + col;
            const float4 p0 = *(const float4*)(gp);
            const float4 p1 = *(const float4*)(gp + 4);
            const float xs[8] = { p0.x, p0.y, p0.z, p0.w, p1.x, p1.y, p1.z, p1.w };
            v8bf hv, lv;
            #pragma unroll
            for (int i = 0; i < 8; ++i) {
                const float  x = xs[i];
                const __bf16 h = (__bf16)x;
                hv[i] = h;
                lv[i] = (__bf16)(x - (float)h);
            }
            *(v8bf*)&AsH[row * RS + col] = hv;
            *(v8bf*)&AsL[row * RS + col] = lv;
        }
        __syncthreads();

        #pragma unroll
        for (int kk = 0; kk < KC; kk += 32) {
            bffrag bh, bl, a0h, a0l, a1h, a1l;
            bh.h[0] = *(const v8bf*)(bhp + kc + kk);        // global_load_b128
            bh.h[1] = *(const v8bf*)(bhp + kc + kk + 16);
            bl.h[0] = *(const v8bf*)(blp + kc + kk);
            bl.h[1] = *(const v8bf*)(blp + kc + kk + 16);
            a0h.h[0] = *(const v8bf*)(a0hp + kk);           // ds_load_b128
            a0h.h[1] = *(const v8bf*)(a0hp + kk + 16);
            a0l.h[0] = *(const v8bf*)(a0lp + kk);
            a0l.h[1] = *(const v8bf*)(a0lp + kk + 16);
            a1h.h[0] = *(const v8bf*)(a1hp + kk);
            a1h.h[1] = *(const v8bf*)(a1hp + kk + 16);
            a1l.h[0] = *(const v8bf*)(a1lp + kk);
            a1l.h[1] = *(const v8bf*)(a1lp + kk + 16);

            // token tiles 0 (c0) and 1 (c1): hi*hi + hi*lo + lo*hi (fp32 accumulate)
            c0 = __builtin_amdgcn_wmma_f32_16x16x32_bf16(
                     false, a0h.v, false, bh.v, (short)0, c0, false, false);
            c1 = __builtin_amdgcn_wmma_f32_16x16x32_bf16(
                     false, a1h.v, false, bh.v, (short)0, c1, false, false);
            c0 = __builtin_amdgcn_wmma_f32_16x16x32_bf16(
                     false, a0h.v, false, bl.v, (short)0, c0, false, false);
            c1 = __builtin_amdgcn_wmma_f32_16x16x32_bf16(
                     false, a1h.v, false, bl.v, (short)0, c1, false, false);
            c0 = __builtin_amdgcn_wmma_f32_16x16x32_bf16(
                     false, a0l.v, false, bh.v, (short)0, c0, false, false);
            c1 = __builtin_amdgcn_wmma_f32_16x16x32_bf16(
                     false, a1l.v, false, bh.v, (short)0, c1, false, false);
        }
        __syncthreads();
    }

    // --- epilogue: write logits, compute sigmoid scores into LDS
    {
        const int rbase = (lane >> 4) * 8;   // C layout: VGPR i, lanes>=16 -> row i+8
        #pragma unroll
        for (int i = 0; i < 8; ++i) {
            const int row0 = rbase + i;            // token tile 0
            const int row1 = row0 + 16;            // token tile 1
            const float x0 = c0[i];
            const float x1 = c1[i];
            out_logits[(size_t)(token0 + row0) * NUM_EXPERTS + expert] = x0;
            out_logits[(size_t)(token0 + row1) * NUM_EXPERTS + expert] = x1;
            Sc[row0 * SC_STRIDE + expert] = 1.0f / (1.0f + __expf(-x0));
            Sc[row1 * SC_STRIDE + expert] = 1.0f / (1.0f + __expf(-x1));
        }
    }
    __syncthreads();

    // --- routing: one token per lane on the first 32 threads (serial, tiny)
    if (tid < TOK_TILE) {
        const float* sc = &Sc[tid * SC_STRIDE];

        // group scores = sum of top-2 (score + bias) within each group of 32
        float gsc[N_GROUP];
        #pragma unroll
        for (int g = 0; g < N_GROUP; ++g) {
            float m1 = -1e30f, m2 = -1e30f;
            const int e0 = g * EPG;
            for (int e = e0; e < e0 + EPG; ++e) {
                const float s = sc[e] + Bl[e];
                if (s > m1)      { m2 = m1; m1 = s; }
                else if (s > m2) { m2 = s; }
            }
            gsc[g] = m1 + m2;
        }

        // pick top-4 groups (strict > : ties resolve to lowest index, like jax)
        bool gm[N_GROUP];
        #pragma unroll
        for (int g = 0; g < N_GROUP; ++g) gm[g] = false;
        for (int r = 0; r < TOPK_GROUP; ++r) {
            int bi = 0; float bv = -1e30f;
            #pragma unroll
            for (int g = 0; g < N_GROUP; ++g)
                if (!gm[g] && gsc[g] > bv) { bv = gsc[g]; bi = g; }
            gm[bi] = true;
        }

        // top-8 experts across the 4 selected groups (insertion, stable ties)
        float tv[TOP_K]; int ti[TOP_K];
        #pragma unroll
        for (int j = 0; j < TOP_K; ++j) { tv[j] = -1e30f; ti[j] = 0; }
        for (int e = 0; e < NUM_EXPERTS; ++e) {
            if (!gm[e >> 5]) continue;
            const float s = sc[e] + Bl[e];
            if (s > tv[TOP_K - 1]) {
                int j = TOP_K - 1;
                while (j > 0 && s > tv[j - 1]) { tv[j] = tv[j - 1]; ti[j] = ti[j - 1]; --j; }
                tv[j] = s; ti[j] = e;
            }
        }

        // weights = raw sigmoid scores at selected experts, normalized, scaled
        float w[TOP_K]; float wsum = 0.0f;
        #pragma unroll
        for (int j = 0; j < TOP_K; ++j) { w[j] = sc[ti[j]]; wsum += w[j]; }
        const float inv = ROUTED_SCALING / (wsum + 1e-20f);

        const size_t tok = (size_t)(token0 + tid);
        #pragma unroll
        for (int j = 0; j < TOP_K; ++j) {
            out_idx[tok * TOP_K + j] = (float)ti[j];
            out_w  [tok * TOP_K + j] = w[j] * inv;
        }
    }
}

extern "C" void kernel_launch(void* const* d_in, const int* in_sizes, int n_in,
                              void* d_out, int out_size, void* d_ws, size_t ws_size,
                              hipStream_t stream) {
    const float* hs     = (const float*)d_in[0];   // (N, 2048) f32
    const float* weight = (const float*)d_in[1];   // (256, 2048) f32
    const float* bias   = (const float*)d_in[2];   // (256,) f32

    const int n_tokens = in_sizes[0] / HIDDEN;     // 32768
    const int n_w      = in_sizes[1];              // 256*2048

    // workspace: bf16 hi/lo planes of the weight matrix (2 MB total)
    __bf16* w_hi = (__bf16*)d_ws;
    __bf16* w_lo = w_hi + (size_t)n_w;

    // d_out: tuple (topk_idx, topk_weight, logits) flattened in return order
    float* out        = (float*)d_out;
    float* out_idx    = out;                                   // N*8
    float* out_w      = out + (size_t)n_tokens * TOP_K;        // N*8
    float* out_logits = out + (size_t)n_tokens * TOP_K * 2;    // N*256

    split_weight_kernel<<<(n_w + 511) / 512, 512, 0, stream>>>(weight, w_hi, w_lo, n_w);

    const int grid = n_tokens / TOK_TILE;                      // 1024 blocks
    moe_gate_kernel<<<grid, 512, 0, stream>>>(hs, w_hi, w_lo, bias,
                                              out_idx, out_w, out_logits);
}
